// YOLOXHead2_17282948399258
// MI455X (gfx1250) — compile-verified, matrix operand order checked by reading, not run on hardware
//
#include <hip/hip_runtime.h>
#include <hip/hip_bf16.h>

// ---------------------------------------------------------------------------
// YOLOX head for MI455X (gfx1250, wave32, WMMA).
// Implicit-GEMM convolutions on v_wmma_f32_16x16x32_f16, double-buffered LDS,
// hoisted A-tile address math, async global->LDS weight staging when available.
// ---------------------------------------------------------------------------

typedef __attribute__((ext_vector_type(16))) _Float16 v16h;
typedef __attribute__((ext_vector_type(8)))  _Float16 v8h;
typedef __attribute__((ext_vector_type(8)))  float    v8f;
typedef __attribute__((ext_vector_type(4)))  int      v4i;

union V16 { v16h v; v8h h[2]; };

#define BATCH      4
#define OUT_ROWS   20160      // 15360 + 3840 + 960 per batch
#define OUT_C      262        // 4 + 1 + 1 + 256
#define BN_EPS     1e-5f

#if defined(__AMDGCN__) && __has_builtin(__builtin_amdgcn_global_load_async_to_lds_b128)
#define USE_ASYNC_B 1
// async builtin signature (from hipcc diagnostics): 
//   (AS1 v4i* src, AS3 v4i* dst, int offset, int cpol)
typedef __attribute__((address_space(1))) v4i* g_v4i_p;
typedef __attribute__((address_space(3))) v4i* l_v4i_p;
#else
#define USE_ASYNC_B 0
#endif

// ---------------------------------------------------------------------------
// Weight packing: fold BN into weights, convert to f16, layout [co][tap][ci]
// so a K-chunk of 32 is 32 contiguous input channels of one 3x3 tap.
// ---------------------------------------------------------------------------
__global__ __launch_bounds__(256)
void pack_weights(const float* __restrict__ w,
                  const float* __restrict__ bn_g, const float* __restrict__ bn_b,
                  const float* __restrict__ bn_m, const float* __restrict__ bn_v,
                  _Float16* __restrict__ wp, float* __restrict__ bias,
                  int cout, int cin, int kk, int hasBN)
{
    int idx = blockIdx.x * blockDim.x + threadIdx.x;
    int total = cout * cin * kk;
    if (idx >= total) return;
    int tap = idx % kk;
    int ci  = (idx / kk) % cin;
    int co  = idx / (kk * cin);
    float scale = 1.f, bs;
    if (hasBN) {
        scale = bn_g[co] * rsqrtf(bn_v[co] + BN_EPS);
        bs    = bn_b[co] - bn_m[co] * scale;
    } else {
        bs = bn_b[co];
    }
    wp[(size_t)co * (kk * cin) + tap * cin + ci] = (_Float16)(w[idx] * scale);
    if (ci == 0 && tap == 0) bias[co] = bs;
}

// ---------------------------------------------------------------------------
// Implicit-GEMM conv. Block = 256 threads (8 waves). Tile: M=128 pixels,
// N=64 output channels. Wave (wm 0..3, wn 0..1) computes a 32x32 patch as
// 2x2 WMMA 16x16 tiles, K stepped by 32.
//
// A-tile staging: thread t owns fixed pixel row ar = t & 127 and 16
// contiguous channels cb..cb+15 (cb = (t>>7)*16).  y/x/tap-validity and the
// global base address are hoisted out of the K-loop; per chunk only a uniform
// scalar delta (ci0*P + dy*W + dx) is added.  LDS side: 2x ds_store_b128.
//
//   AF16  : input activations are f16 (else f32, for the stem)
//   K3    : 3x3 pad-1 conv (else 1x1)
//   RAWOUT: f32 bias-added result strided into [B,20160,262] at chan 6
//           (reid_pred); else SiLU -> f16 NCHW act buffer
// ---------------------------------------------------------------------------
struct GemmArgs {
    const void*     A;      // activations (f32 or f16, NCHW [B][cin][P])
    const _Float16* W;      // packed weights [cout][Ktot]
    const float*    bias;   // [cout]
    void*           out;    // f16 act buffer or f32 final output
    int cin, H, Wd, P, Ktot, rowoff, cout;
};

template<bool AF16, bool K3, bool RAWOUT>
__global__ __launch_bounds__(256)
void conv_gemm(GemmArgs a)
{
    __shared__ __align__(16) _Float16 AL[2][128 * 40];  // 128 M x 32 K (pad->40)
    __shared__ __align__(16) _Float16 BL[2][64 * 40];   //  64 N x 32 K (pad->40)

    const int tid  = threadIdx.x;
    const int b    = blockIdx.z;
    const int p0   = blockIdx.x * 128;
    const int co0  = blockIdx.y * 64;
    const int wid  = tid >> 5;
    const int lane = tid & 31;
    const int wm   = wid >> 1;          // 0..3 : M quadrant
    const int wn   = wid & 1;           // 0..1 : N half
    const int half = lane >> 4;         // K split per 16-bit A-layout
    const int lrow = lane & 15;

    // ---- hoisted A staging state (constant across the K loop) -------------
    const int ar = tid & 127;           // pixel row within tile
    const int cb = (tid >> 7) << 4;     // first of 16 contiguous channels
    const int p  = p0 + ar;
    int y = 0, x = 0;
    unsigned tapmask = 0u;              // bit t = tap t in-bounds for this pixel
    if (p < a.P) {
        y = p / a.Wd;
        x = p - y * a.Wd;
        if (K3) {
            #pragma unroll
            for (int t9 = 0; t9 < 9; ++t9) {
                int dy = t9 / 3 - 1, dx = t9 % 3 - 1;
                if ((unsigned)(y + dy) < (unsigned)a.H &&
                    (unsigned)(x + dx) < (unsigned)a.Wd)
                    tapmask |= 1u << t9;
            }
        } else {
            tapmask = 1u;
        }
    }
    const long abase = (long)(b * a.cin + cb) * a.P + (long)y * a.Wd + x;

    // ---- hoisted B staging state ------------------------------------------
    const int seg  = tid & 3;           // 8-half segment within a K row
    const int brow = tid >> 2;          // output channel row 0..63
    const _Float16* wptr = a.W + (size_t)(co0 + brow) * a.Ktot + (seg << 3);

    v8f acc[2][2] = {};

    const int nkc = a.Ktot >> 5;        // K chunks of 32
    for (int kc = 0; kc < nkc; ++kc) {
        const int buf = kc & 1;
        int tap, ci0;
        if (K3) { tap = kc >> 3; ci0 = (kc & 7) << 5; }   // cin==256 for 3x3
        else    { tap = 0;       ci0 = kc << 5; }
        const int dy = tap / 3 - 1;
        const int dx = tap % 3 - 1;
        const long delta = (long)ci0 * a.P + (K3 ? (dy * a.Wd + dx) : 0);
        const bool ok = (tapmask >> tap) & 1u;

        // ---- stage A: 16 channels for this thread's pixel -----------------
        v8h alo, ahi;
        #pragma unroll
        for (int i = 0; i < 8; ++i) { alo[i] = (_Float16)0.f; ahi[i] = (_Float16)0.f; }
        if (ok) {
            if (AF16) {
                const _Float16* src = (const _Float16*)a.A + abase + delta;
                #pragma unroll
                for (int i = 0; i < 8; ++i) {
                    alo[i] = src[(size_t)i * a.P];
                    ahi[i] = src[(size_t)(i + 8) * a.P];
                }
            } else {
                const float* src = (const float*)a.A + abase + delta;
                #pragma unroll
                for (int i = 0; i < 8; ++i) {
                    alo[i] = (_Float16)src[(size_t)i * a.P];
                    ahi[i] = (_Float16)src[(size_t)(i + 8) * a.P];
                }
            }
        }
        *(v8h*)&AL[buf][ar * 40 + cb]     = alo;
        *(v8h*)&AL[buf][ar * 40 + cb + 8] = ahi;

        // ---- stage B: one 16-byte chunk per thread ------------------------
#if USE_ASYNC_B
        __builtin_amdgcn_global_load_async_to_lds_b128(
            (g_v4i_p)(wptr + ((size_t)kc << 5)),
            (l_v4i_p)&BL[buf][brow * 40 + (seg << 3)],
            0, 0);
#else
        *(v8h*)&BL[buf][brow * 40 + (seg << 3)] =
            *(const v8h*)(wptr + ((size_t)kc << 5));
#endif

#if USE_ASYNC_B
#if __has_builtin(__builtin_amdgcn_s_wait_asynccnt)
        __builtin_amdgcn_s_wait_asynccnt(0);
#else
        asm volatile("s_wait_asynccnt 0" ::: "memory");
#endif
#endif
        __syncthreads();   // double-buffered: single barrier per K-step

        // ---- fragments per 16-bit 16x32 A layout --------------------------
        // lanes 0-15: K 0..7 / 16..23 ; lanes 16-31: K 8..15 / 24..31
        V16 af[2], bf[2];
        #pragma unroll
        for (int mt = 0; mt < 2; ++mt) {
            int r = wm * 32 + mt * 16 + lrow;
            af[mt].h[0] = *(const v8h*)&AL[buf][r * 40 + half * 8];
            af[mt].h[1] = *(const v8h*)&AL[buf][r * 40 + half * 8 + 16];
        }
        #pragma unroll
        for (int nt = 0; nt < 2; ++nt) {
            int r = wn * 32 + nt * 16 + lrow;
            bf[nt].h[0] = *(const v8h*)&BL[buf][r * 40 + half * 8];
            bf[nt].h[1] = *(const v8h*)&BL[buf][r * 40 + half * 8 + 16];
        }
        #pragma unroll
        for (int mt = 0; mt < 2; ++mt)
            #pragma unroll
            for (int nt = 0; nt < 2; ++nt)
                acc[mt][nt] = __builtin_amdgcn_wmma_f32_16x16x32_f16(
                    false, af[mt].v, false, bf[nt].v,
                    (short)0, acc[mt][nt], false, false);
    }

    // ---- epilogue: C layout VGPR j -> M=(half*8+j), N=lane&15 -------------
    #pragma unroll
    for (int mt = 0; mt < 2; ++mt) {
        #pragma unroll
        for (int nt = 0; nt < 2; ++nt) {
            int co = co0 + wn * 32 + nt * 16 + lrow;
            float bv = a.bias[co];
            #pragma unroll
            for (int j = 0; j < 8; ++j) {
                int m = p0 + wm * 32 + mt * 16 + half * 8 + j;
                if (m >= a.P) continue;
                float val = acc[mt][nt][j] + bv;
                if (RAWOUT) {
                    ((float*)a.out)[((size_t)b * OUT_ROWS + a.rowoff + m) * OUT_C + 6 + co] = val;
                } else {
                    float s = val / (1.f + __expf(-val));   // SiLU
                    ((_Float16*)a.out)[((size_t)(b * a.cout + co)) * a.P + m] = (_Float16)s;
                }
            }
        }
    }
}

// ---------------------------------------------------------------------------
// Small 1x1 preds (6 of 262 channels, <2% of FLOPs) + YOLOX decode.
// ---------------------------------------------------------------------------
__global__ __launch_bounds__(256)
void pred_reg_kernel(const _Float16* __restrict__ regf,
                     const float* __restrict__ regw, const float* __restrict__ regb,
                     const float* __restrict__ objw, const float* __restrict__ objb,
                     float* __restrict__ out, int P, int Wd, int rowoff, int stride)
{
    int idx = blockIdx.x * blockDim.x + threadIdx.x;
    if (idx >= BATCH * P) return;
    int b = idx / P, p = idx - b * P;
    float a0 = 0.f, a1 = 0.f, a2 = 0.f, a3 = 0.f, ao = 0.f;
    for (int ci = 0; ci < 256; ++ci) {
        float f = (float)regf[((size_t)(b * 256 + ci)) * P + p];
        a0 += f * regw[ci];
        a1 += f * regw[256 + ci];
        a2 += f * regw[512 + ci];
        a3 += f * regw[768 + ci];
        ao += f * objw[ci];
    }
    a0 += regb[0]; a1 += regb[1]; a2 += regb[2]; a3 += regb[3]; ao += objb[0];
    int y = p / Wd, x = p - y * Wd;
    float* o = out + ((size_t)b * OUT_ROWS + rowoff + p) * OUT_C;
    o[0] = (a0 + (float)x) * (float)stride;
    o[1] = (a1 + (float)y) * (float)stride;
    o[2] = __expf(a2) * (float)stride;
    o[3] = __expf(a3) * (float)stride;
    o[4] = 1.f / (1.f + __expf(-ao));
}

__global__ __launch_bounds__(256)
void pred_cls_kernel(const _Float16* __restrict__ clsf,
                     const float* __restrict__ clsw, const float* __restrict__ clsb,
                     float* __restrict__ out, int P, int rowoff)
{
    int idx = blockIdx.x * blockDim.x + threadIdx.x;
    if (idx >= BATCH * P) return;
    int b = idx / P, p = idx - b * P;
    float acc = 0.f;
    for (int ci = 0; ci < 256; ++ci)
        acc += (float)clsf[((size_t)(b * 256 + ci)) * P + p] * clsw[ci];
    acc += clsb[0];
    out[((size_t)b * OUT_ROWS + rowoff + p) * OUT_C + 5] = 1.f / (1.f + __expf(-acc));
}

// ---------------------------------------------------------------------------
// Host launcher.
// Input flattening assumed (jax tree order, dict keys alphabetical):
//   d_in[0..128]  = params (43 arrays per level, levels in list order)
//   d_in[129..131]= x0, x1, x2
// Per level base b = l*43:
//   b+0..4   cls_convs[0]  (bn.b, bn.g, bn.m, bn.v, w)
//   b+5..9   cls_convs[1]
//   b+10,11  cls_pred      (b, w)
//   b+12,13  obj_pred      (b, w)
//   b+14..18 reg_convs[0]   b+19..23 reg_convs[1]
//   b+24,25  reg_pred
//   b+26..30 reid_convs[0]  b+31..35 reid_convs[1]
//   b+36,37  reid_pred
//   b+38..42 stem          (bn.b, bn.g, bn.m, bn.v, w)
// ---------------------------------------------------------------------------
extern "C" void kernel_launch(void* const* d_in, const int* in_sizes, int n_in,
                              void* d_out, int out_size, void* d_ws, size_t ws_size,
                              hipStream_t stream)
{
    const int H[3]    = {96, 48, 24};
    const int Wd[3]   = {160, 80, 40};
    const int CIN[3]  = {256, 512, 1024};
    const int STR[3]  = {8, 16, 32};
    const int ROFF[3] = {0, 15360, 19200};

    char* ws = (char*)d_ws;
    const size_t SZ_ACT = (size_t)BATCH * 256 * 15360 * sizeof(_Float16); // 31.5 MB
    _Float16* S    = (_Float16*)(ws);
    _Float16* T1   = (_Float16*)(ws + SZ_ACT);
    _Float16* T2   = (_Float16*)(ws + 2 * SZ_ACT);
    _Float16* WP   = (_Float16*)(ws + 3 * SZ_ACT);                 // max 256*2304 f16
    float*    BIAS = (float*)   (ws + 3 * SZ_ACT + (size_t)256 * 2304 * sizeof(_Float16));
    float*    out  = (float*)d_out;

    for (int l = 0; l < 3; ++l) {
        const int base = l * 43;
        const int P = H[l] * Wd[l];
        const float* x = (const float*)d_in[129 + l];
        dim3 grid((P + 127) / 128, 256 / 64, BATCH);
        int npx = (BATCH * P + 255) / 256;

        auto F = [&](int i) { return (const float*)d_in[i]; };

        auto packBN = [&](int pb, int wi, int cout, int cin, int kk) {
            int tot = cout * cin * kk;
            pack_weights<<<(tot + 255) / 256, 256, 0, stream>>>(
                F(wi), F(pb + 1), F(pb + 0), F(pb + 2), F(pb + 3),
                WP, BIAS, cout, cin, kk, 1);
        };
        auto conv3 = [&](int pb, const _Float16* in, _Float16* ob) {
            packBN(pb, pb + 4, 256, 256, 9);
            GemmArgs g{in, WP, BIAS, ob, 256, H[l], Wd[l], P, 2304, 0, 256};
            conv_gemm<true, true, false><<<grid, 256, 0, stream>>>(g);
        };

        // ---- stem: 1x1 cin -> 256, f32 in, SiLU, f16 out ------------------
        packBN(base + 38, base + 42, 256, CIN[l], 1);
        {
            GemmArgs g{x, WP, BIAS, S, CIN[l], H[l], Wd[l], P, CIN[l], 0, 256};
            conv_gemm<false, false, false><<<grid, 256, 0, stream>>>(g);
        }

        // ---- cls branch ---------------------------------------------------
        conv3(base + 0, S, T1);
        conv3(base + 5, T1, T2);
        pred_cls_kernel<<<npx, 256, 0, stream>>>(T2, F(base + 11), F(base + 10),
                                                 out, P, ROFF[l]);

        // ---- reg branch ---------------------------------------------------
        conv3(base + 14, S, T1);
        conv3(base + 19, T1, T2);
        pred_reg_kernel<<<npx, 256, 0, stream>>>(T2, F(base + 25), F(base + 24),
                                                 F(base + 13), F(base + 12),
                                                 out, P, Wd[l], ROFF[l], STR[l]);

        // ---- reid branch (256-ch pred via WMMA, strided f32 epilogue) -----
        conv3(base + 26, S, T1);
        conv3(base + 31, T1, T2);
        {
            int tot = 256 * 256;
            pack_weights<<<(tot + 255) / 256, 256, 0, stream>>>(
                F(base + 37), nullptr, F(base + 36), nullptr, nullptr,
                WP, BIAS, 256, 256, 1, 0);
            GemmArgs g{T2, WP, BIAS, out, 256, H[l], Wd[l], P, 256, ROFF[l], 256};
            conv_gemm<true, false, true><<<grid, 256, 0, stream>>>(g);
        }
    }
}
